// Transformer_47545287966777
// MI455X (gfx1250) — matrix operand order repaired and checked by reading, not compiled, not run on hardware
//
#include <hip/hip_runtime.h>
#include <hip/hip_bf16.h>
#include <stdint.h>

// ---------------------------------------------------------------------------
// TransformerConv (heads=1) x3 layers for MI455X / gfx1250.
//   Per layer:
//     1. convert h (f32) -> hh (f16)           [+ReLU for layers 1,2 inputs]
//     2. WMMA GEMM: q,k,v = hh@W, ho = hh@Ws+bs (f32 accum, f16 inputs)
//        - LDS tiles filled by the Tensor Data Mover (tensor_load_to_lds)
//     3. edge pass A: alpha_e = <q[dst],k[src]>/8, segment-max via atomicMax
//     4. edge pass B: e = exp(alpha-m[dst]), segment-sum via atomic f32 add
//     5. edge pass C: ho[dst] += (e/denom[dst]) * v[src]  (atomic f32 adds)
// ---------------------------------------------------------------------------

typedef _Float16 v16h __attribute__((ext_vector_type(16)));
typedef float    v8f  __attribute__((ext_vector_type(8)));

#define DD 64   // channels
#define LL 3    // layers

// ------------------------------ TDM support --------------------------------
#if defined(__has_builtin)
# if __has_builtin(__builtin_amdgcn_tensor_load_to_lds) && __has_builtin(__builtin_amdgcn_s_wait_tensorcnt)
#  define USE_TDM 1
# endif
#endif
#ifndef USE_TDM
# define USE_TDM 0
#endif

#if USE_TDM
typedef unsigned int u32x4 __attribute__((ext_vector_type(4)));
typedef int          i32x8 __attribute__((ext_vector_type(8)));
typedef int          i32x4 __attribute__((ext_vector_type(4)));

// Pack a 2D (or 1D) f16 tile descriptor per CDNA5 ISA 8.3/8.4 and issue the
// TDM load. Issued once per wave (EXEC ignored by TENSOR ops).
__device__ __forceinline__ void tdm_load_f16(
    unsigned lds_addr, const void* gptr,
    unsigned tensor_d0, unsigned tensor_d1,
    unsigned tile_d0, unsigned tile_d1,
    unsigned long long stride0)
{
  unsigned long long ga = (unsigned long long)(uintptr_t)gptr;
  u32x4 g0;
  g0[0] = 1u;                                   // count=1 (valid), user mode
  g0[1] = lds_addr;                             // lds_addr [63:32]
  g0[2] = (unsigned)ga;                         // global_addr [95:64]
  g0[3] = (unsigned)((ga >> 32) & 0x01FFFFFFu)  // global_addr [120:96]
        | 0x80000000u;                          // type=2 ("image") [127:126]
  i32x8 g1;
  g1[0] = 1 << 16;                              // data_size=1 (2 bytes), mask=0
  g1[1] = (int)((tensor_d0 & 0xFFFFu) << 16);   // tensor_dim0[15:0] @ bit48
  g1[2] = (int)((tensor_d0 >> 16) | ((tensor_d1 & 0xFFFFu) << 16));
  g1[3] = (int)((tensor_d1 >> 16) | ((tile_d0 & 0xFFFFu) << 16));
  g1[4] = (int)(tile_d1 & 0xFFFFu);             // tile_dim1 ; tile_dim2=0
  g1[5] = (int)(unsigned)stride0;               // tensor_dim0_stride[31:0]
  g1[6] = (int)((unsigned)(stride0 >> 32) & 0xFFFFu);
  g1[7] = 0;
  i32x4 z4 = {0, 0, 0, 0};
#if __clang_major__ >= 23
  i32x8 z8 = {0, 0, 0, 0, 0, 0, 0, 0};
  __builtin_amdgcn_tensor_load_to_lds(g0, g1, z4, z4, z8, 0);
#else
  __builtin_amdgcn_tensor_load_to_lds(g0, g1, z4, z4, 0);
#endif
}
#endif  // USE_TDM

// ------------------------- small helper kernels ----------------------------

// f32 -> f16 (optionally fused ReLU of previous layer output)
__global__ void convert_relu_kernel(const float* __restrict__ h,
                                    _Float16* __restrict__ hh,
                                    int n, int do_relu) {
  int i = blockIdx.x * blockDim.x + threadIdx.x;
  if (i >= n) return;
  float x = h[i];
  if (do_relu) x = fmaxf(x, 0.0f);
  hh[i] = (_Float16)x;
}

// Transpose + convert all weights once: WT[l][mat][n][k] = W_mat[l][k][n]
__global__ void prep_weights_kernel(const float* __restrict__ Wq,
                                    const float* __restrict__ Wk,
                                    const float* __restrict__ Wv,
                                    const float* __restrict__ Ws,
                                    _Float16* __restrict__ WT) {
  int i = blockIdx.x * blockDim.x + threadIdx.x;
  if (i >= LL * 4 * DD * DD) return;
  int k   = i & 63;
  int n   = (i >> 6) & 63;
  int mat = (i >> 12) & 3;
  int l   = i >> 14;
  const float* W = (mat == 0) ? Wq : (mat == 1) ? Wk : (mat == 2) ? Wv : Ws;
  WT[i] = (_Float16)W[(l * DD + k) * DD + n];
}

__global__ void zero_md_kernel(unsigned* __restrict__ menc,
                               float* __restrict__ denom, int n) {
  int i = blockIdx.x * blockDim.x + threadIdx.x;
  if (i < n) { menc[i] = 0u; denom[i] = 0.0f; }
}

// monotonic float <-> uint encoding so atomicMax(uint) == float max
__device__ __forceinline__ unsigned f2ord(float f) {
  unsigned u = __float_as_uint(f);
  return (u & 0x80000000u) ? ~u : (u | 0x80000000u);
}
__device__ __forceinline__ float ord2f(unsigned u) {
  return (u & 0x80000000u) ? __uint_as_float(u & 0x7FFFFFFFu)
                           : __uint_as_float(~u);
}

__device__ __forceinline__ void atom_add_f32(float* p, float v) {
  unsafeAtomicAdd(p, v);   // lowers to global_atomic_add_f32
}

// ------------------------------ WMMA GEMM ----------------------------------
// Block: 256 threads = 8 waves, 32 node-rows per block.
// Wave w: mt = w&1 selects 16-row half; mat = w>>1 selects {q,k,v,skip}.
// Each wave: 16x64 output tile = 4 N-tiles, K loop of 2 steps of K=32.
__global__ __launch_bounds__(256) void gemm_qkvs_kernel(
    const _Float16* __restrict__ hh,   // [N][64] f16
    const _Float16* __restrict__ WT,   // [4][64][64] f16, [mat][n][k]
    const float* __restrict__ bq, const float* __restrict__ bk,
    const float* __restrict__ bv, const float* __restrict__ bs,
    float* __restrict__ q, float* __restrict__ k, float* __restrict__ v,
    float* __restrict__ ho, int nrows) {
  __shared__ _Float16 xs[32][DD];          // 4 KB  x-tile
  __shared__ _Float16 ws_[4][DD][DD];      // 32 KB transposed weights

  const int tid  = threadIdx.x;
  const int wave = tid >> 5;
  const int lane = tid & 31;
  const int l16  = lane & 15;
  const int half = lane >> 4;
  const int rowBase = blockIdx.x * 32;

#if USE_TDM
  if (wave == 0) {
    // x tile: 2D, 64 elems/row, 32 rows
    tdm_load_f16((unsigned)(uintptr_t)&xs[0][0],
                 hh + (size_t)rowBase * DD,
                 DD, (unsigned)nrows, DD, 32, DD);
    // weights: flat 1D, 4*64*64 f16
    tdm_load_f16((unsigned)(uintptr_t)&ws_[0][0][0],
                 WT, 4 * DD * DD, 1, 4 * DD * DD, 1, 4 * DD * DD);
    __builtin_amdgcn_s_wait_tensorcnt(0);
  }
  __syncthreads();
#else
  for (int i = tid; i < 32 * DD; i += 256) {
    int r = rowBase + (i >> 6);
    xs[0][i] = (r < nrows) ? hh[(size_t)r * DD + (i & 63)] : (_Float16)0.0f;
  }
  for (int i = tid; i < 4 * DD * DD; i += 256) ws_[0][0][i] = WT[i];
  __syncthreads();
#endif

  const int mat = wave >> 1;
  const int mt  = wave & 1;
  const float* bptr = (mat == 0) ? bq : (mat == 1) ? bk : (mat == 2) ? bv : bs;
  float* optr       = (mat == 0) ? q  : (mat == 1) ? k  : (mat == 2) ? v  : ho;

  v8f acc[4];
#pragma unroll
  for (int nt = 0; nt < 4; ++nt) {
    float b = bptr[nt * 16 + l16];
#pragma unroll
    for (int r = 0; r < 8; ++r) acc[nt][r] = b;
  }

#pragma unroll
  for (int ks = 0; ks < 2; ++ks) {
    // A fragment: lane = row (M = mt*16 + l16); elements per ISA 7.12.2:
    // e<8 -> K = ks*32 + 8*half + e ; e>=8 -> +16
    const _Float16* abase = &xs[mt * 16 + l16][ks * 32 + 8 * half];
    v16h A;
#pragma unroll
    for (int e = 0; e < 8; ++e) { A[e] = abase[e]; A[8 + e] = abase[16 + e]; }

#pragma unroll
    for (int nt = 0; nt < 4; ++nt) {
      // B fragment: lane = col (N = nt*16 + l16); e<8 -> K = ks*32 + 8*half + e ; e>=8 -> +16
      // ws_ holds W^T, so this is two contiguous 16-byte LDS runs.
      const _Float16* bbase = &ws_[mat][nt * 16 + l16][ks * 32 + 8 * half];
      v16h B;
#pragma unroll
      for (int e = 0; e < 8; ++e) { B[e] = bbase[e]; B[8 + e] = bbase[16 + e]; }
      acc[nt] = __builtin_amdgcn_wmma_f32_16x16x32_f16(
          false, A, false, B, (short)0, acc[nt], false, false);
    }
  }

  // C/D layout: VGPR r, lane -> M = r + 8*half, N = l16
  const int row0 = rowBase + mt * 16;
#pragma unroll
  for (int nt = 0; nt < 4; ++nt) {
#pragma unroll
    for (int r = 0; r < 8; ++r) {
      int row = row0 + r + 8 * half;
      if (row < nrows) optr[(size_t)row * DD + nt * 16 + l16] = acc[nt][r];
    }
  }
}

// --------------------------- edge-phase kernels ----------------------------

// alpha_e = <q[dst], k[src]> / sqrt(64); segment-max into menc (ordered uint)
__global__ __launch_bounds__(256) void edge_logits_kernel(
    const float4* __restrict__ q4, const float4* __restrict__ k4,
    const int* __restrict__ src, const int* __restrict__ dst,
    float* __restrict__ alpha, unsigned* __restrict__ menc, int nE) {
  int t = blockIdx.x * 256 + threadIdx.x;
  int e = t >> 4, j = t & 15;
  if (e >= nE) return;
  int s = src[e], d = dst[e];
  float4 qv = q4[(size_t)d * 16 + j];
  float4 kv = k4[(size_t)s * 16 + j];
  float p = qv.x * kv.x + qv.y * kv.y + qv.z * kv.z + qv.w * kv.w;
  p += __shfl_xor(p, 8);
  p += __shfl_xor(p, 4);
  p += __shfl_xor(p, 2);
  p += __shfl_xor(p, 1);
  if (j == 0) {
    float a = p * 0.125f;   // 1/sqrt(64)
    alpha[e] = a;
    atomicMax(&menc[d], f2ord(a));
  }
}

// e = exp(alpha - m[dst]); denom[dst] += e   (alpha overwritten in place)
__global__ __launch_bounds__(256) void edge_exp_kernel(
    const int* __restrict__ dst, const unsigned* __restrict__ menc,
    float* __restrict__ alpha, float* __restrict__ denom, int nE) {
  int e = blockIdx.x * 256 + threadIdx.x;
  if (e >= nE) return;
  int d = dst[e];
  float ex = __expf(alpha[e] - ord2f(menc[d]));
  alpha[e] = ex;
  atom_add_f32(&denom[d], ex);
}

// ho[dst] += (e / (denom[dst]+1e-16)) * v[src]
__global__ __launch_bounds__(256) void edge_agg_kernel(
    const float* __restrict__ alpha, const float* __restrict__ denom,
    const float4* __restrict__ v4, const int* __restrict__ src,
    const int* __restrict__ dst, float* __restrict__ ho, int nE) {
  int t = blockIdx.x * 256 + threadIdx.x;
  int e = t >> 4, j = t & 15;
  if (e >= nE) return;
  int s = src[e], d = dst[e];
  float a = alpha[e] / (denom[d] + 1e-16f);
  float4 vv = v4[(size_t)s * 16 + j];
  float* o = ho + (size_t)d * DD + j * 4;
  atom_add_f32(o + 0, a * vv.x);
  atom_add_f32(o + 1, a * vv.y);
  atom_add_f32(o + 2, a * vv.z);
  atom_add_f32(o + 3, a * vv.w);
}

// ------------------------------- launcher ----------------------------------

static inline size_t align256(size_t x) { return (x + 255) & ~(size_t)255; }

extern "C" void kernel_launch(void* const* d_in, const int* in_sizes, int n_in,
                              void* d_out, int out_size, void* d_ws,
                              size_t ws_size, hipStream_t stream) {
  const float* x  = (const float*)d_in[0];
  const int*   ei = (const int*)d_in[1];
  // d_in[2] = edge_type (unused by reference math)
  const float* Wq = (const float*)d_in[3];
  const float* bq = (const float*)d_in[4];
  const float* Wk = (const float*)d_in[5];
  const float* bk = (const float*)d_in[6];
  const float* Wv = (const float*)d_in[7];
  const float* bv = (const float*)d_in[8];
  const float* Ws = (const float*)d_in[9];
  const float* bs = (const float*)d_in[10];

  const int N = in_sizes[0] / DD;
  const int E = in_sizes[1] / 2;
  const int* src = ei;
  const int* dst = ei + E;

  // workspace carve-up (~123 MB)
  char* w = (char*)d_ws;
  _Float16* WT = (_Float16*)w;  w += align256((size_t)LL * 4 * DD * DD * 2);
  _Float16* hh = (_Float16*)w;  w += align256((size_t)N * DD * 2);
  float* q     = (float*)w;     w += align256((size_t)N * DD * 4);
  float* k     = (float*)w;     w += align256((size_t)N * DD * 4);
  float* v     = (float*)w;     w += align256((size_t)N * DD * 4);
  float* hoBuf = (float*)w;     w += align256((size_t)N * DD * 4);
  float* alpha = (float*)w;     w += align256((size_t)E * 4);
  unsigned* menc = (unsigned*)w; w += align256((size_t)N * 4);
  float* denom = (float*)w;     w += align256((size_t)N * 4);
  (void)ws_size; (void)n_in; (void)out_size;

  prep_weights_kernel<<<(LL * 4 * DD * DD + 255) / 256, 256, 0, stream>>>(
      Wq, Wk, Wv, Ws, WT);

  const float* hin = x;
  float* houts[LL] = {hoBuf, hoBuf, (float*)d_out};

  for (int l = 0; l < LL; ++l) {
    convert_relu_kernel<<<(N * DD + 255) / 256, 256, 0, stream>>>(
        hin, hh, N * DD, l > 0 ? 1 : 0);

    gemm_qkvs_kernel<<<(N + 31) / 32, 256, 0, stream>>>(
        hh, WT + (size_t)l * 4 * DD * DD,
        bq + l * DD, bk + l * DD, bv + l * DD, bs + l * DD,
        q, k, v, houts[l], N);

    zero_md_kernel<<<(N + 255) / 256, 256, 0, stream>>>(menc, denom, N);

    edge_logits_kernel<<<(E * 16 + 255) / 256, 256, 0, stream>>>(
        (const float4*)q, (const float4*)k, src, dst, alpha, menc, E);

    edge_exp_kernel<<<(E + 255) / 256, 256, 0, stream>>>(
        dst, menc, alpha, denom, E);

    edge_agg_kernel<<<(E * 16 + 255) / 256, 256, 0, stream>>>(
        alpha, denom, (const float4*)v, src, dst, houts[l], E);

    hin = houts[l];
  }
}
